// MoELayer_6545530159427
// MI455X (gfx1250) — matrix operand order repaired and checked by reading, not compiled, not run on hardware
//
#include <hip/hip_runtime.h>
#include <hip/hip_bf16.h>

// ---------------- problem dims ----------------
constexpr int HDIM = 1024;
constexpr int EXP  = 8;
constexpr int FDIM = 4096;
constexpr int NTOK = 8192;          // B*S = 4*2048

// ---------------- workspace layout (bytes) ----------------
constexpr size_t OFF_XBF  = 0;                                        // NTOK*HDIM bf16  = 16 MiB
constexpr size_t OFF_W1T  = OFF_XBF + (size_t)NTOK * HDIM * 2;        // E*F*H bf16 (col-major per f) = 64 MiB
constexpr size_t OFF_W2T  = OFF_W1T + (size_t)EXP * HDIM * FDIM * 2;  // E*H*F bf16 (col-major per h) = 64 MiB
constexpr size_t OFF_P0   = OFF_W2T + (size_t)EXP * FDIM * HDIM * 2;  // NTOK f32
constexpr size_t OFF_CNT  = OFF_P0  + (size_t)NTOK * 4;               // EXP ints (+pad)
constexpr size_t OFF_LIST = OFF_CNT + 256;                            // EXP*NTOK ints

// ---------------- vector types ----------------
typedef __attribute__((ext_vector_type(16))) __bf16       v16bf;
typedef __attribute__((ext_vector_type(8)))  float        v8f;
typedef __attribute__((ext_vector_type(4)))  unsigned int u32x4;
typedef __attribute__((ext_vector_type(4)))  int          v4i;

union FragAB { u32x4 u[2]; v16bf v; };

#if defined(__gfx1250__) && __has_builtin(__builtin_amdgcn_global_load_async_to_lds_b128) && \
    __has_builtin(__builtin_amdgcn_s_wait_asynccnt)
#define HAS_ASYNC_LDS 1
#else
#define HAS_ASYNC_LDS 0
#endif

__device__ __forceinline__ float gelu_tanh(float x) {
    float x3 = x * x * x;
    return 0.5f * x * (1.0f + tanhf(0.7978845608028654f * (x + 0.044715f * x3)));
}

// ---------------- kernel: zero output + counters ----------------
__global__ void moe_zero_k(float* out, int n4, int* cnt) {
    int i = blockIdx.x * 256 + threadIdx.x;
    if (i < n4) {
        float4 z = make_float4(0.f, 0.f, 0.f, 0.f);
        *(float4*)(out + (size_t)i * 4) = z;
    }
    if (blockIdx.x == 0 && threadIdx.x < EXP) cnt[threadIdx.x] = 0;
}

// ---------------- kernel: x fp32 -> bf16 ----------------
__global__ void moe_cvtx_k(const float* __restrict__ x, __bf16* __restrict__ xb, int n4) {
    int i = blockIdx.x * 256 + threadIdx.x;
    if (i >= n4) return;
    float4 v = *(const float4*)(x + (size_t)i * 4);
    size_t b = (size_t)i * 4;
    xb[b + 0] = (__bf16)v.x;
    xb[b + 1] = (__bf16)v.y;
    xb[b + 2] = (__bf16)v.z;
    xb[b + 3] = (__bf16)v.w;
}

// ---------------- kernel: fp32 [E][R][C] -> bf16 [E][C][R] (LDS tiled) ----------------
__global__ void moe_transpose_cvt_k(const float* __restrict__ src, __bf16* __restrict__ dst,
                                    int R, int C) {
    int tiles_r = R >> 5, tiles_c = C >> 5;
    int bid = blockIdx.x;
    int e   = bid / (tiles_r * tiles_c);
    int rem = bid % (tiles_r * tiles_c);
    int r0  = (rem / tiles_c) << 5;
    int c0  = (rem % tiles_c) << 5;
    __shared__ float tile[32][33];
    const float* s = src + (size_t)e * R * C;
    __bf16*      d = dst + (size_t)e * R * C;
    int lr = threadIdx.x >> 5, lc = threadIdx.x & 31;
#pragma unroll
    for (int i = 0; i < 4; ++i)
        tile[lr + i * 8][lc] = s[(size_t)(r0 + lr + i * 8) * C + (c0 + lc)];
    __syncthreads();
#pragma unroll
    for (int i = 0; i < 4; ++i)
        d[(size_t)(c0 + lr + i * 8) * R + (r0 + lc)] = (__bf16)tile[lc][lr + i * 8];
}

// ---------------- kernel: router (one wave per token) ----------------
__global__ void moe_router_k(const float* __restrict__ x, const float* __restrict__ Wr,
                             const float* __restrict__ br,
                             float* __restrict__ p0arr, int* __restrict__ cnt,
                             int* __restrict__ list) {
    int wave = threadIdx.x >> 5, lane = threadIdx.x & 31;
    int tok  = blockIdx.x * 8 + wave;
    const float* xr = x + (size_t)tok * HDIM;
    float le[EXP];
#pragma unroll
    for (int i = 0; i < EXP; ++i) le[i] = 0.f;
    for (int h = lane; h < HDIM; h += 32) {
        float xv = xr[h];
        const float* wr = Wr + (size_t)h * EXP;
#pragma unroll
        for (int i = 0; i < EXP; ++i) le[i] += xv * wr[i];
    }
#pragma unroll
    for (int i = 0; i < EXP; ++i)
        for (int off = 16; off; off >>= 1)
            le[i] += __shfl_xor(le[i], off, 32);
    if (lane == 0) {
#pragma unroll
        for (int i = 0; i < EXP; ++i) le[i] += br[i];
        int e0 = 0; float l0 = le[0];
        for (int i = 1; i < EXP; ++i) if (le[i] > l0) { l0 = le[i]; e0 = i; }
        int e1 = -1; float l1 = -3.4e38f;
        for (int i = 0; i < EXP; ++i) if (i != e0 && le[i] > l1) { l1 = le[i]; e1 = i; }
        // normalized top-1 prob (reference applies it to BOTH selected experts)
        float p0 = 1.0f / (1.0f + __expf(l1 - l0));
        p0arr[tok] = p0;
        int i0 = atomicAdd(&cnt[e0], 1); list[e0 * NTOK + i0] = tok;
        int i1 = atomicAdd(&cnt[e1], 1); list[e1 * NTOK + i1] = tok;
    }
}

// ---------------- kernel: fused expert FFN (WMMA bf16) ----------------
// block = 256 threads (8 waves); one block = 16-token tile of one expert.
// wave w owns output columns [w*128, w*128+128) : 8 WMMA n-tiles, acc in VGPRs.
constexpr int XSS = 1032;  // xs row stride in bf16 (padded, 16B aligned)
constexpr int HSS = 264;   // hs row stride in bf16 (padded, 16B aligned)

__global__ void __launch_bounds__(256, 1)
moe_ffn_k(const __bf16* __restrict__ xbf, const __bf16* __restrict__ W1t,
          const __bf16* __restrict__ W2t, const float* __restrict__ b1,
          const float* __restrict__ b2, const float* __restrict__ p0arr,
          const int* __restrict__ cnt, const int* __restrict__ list,
          float* __restrict__ out) {
    int e = blockIdx.x >> 9;
    int t = blockIdx.x & 511;
    int count = cnt[e];
    if (t * 16 >= count) return;

    __shared__ __align__(16) __bf16 xs[16 * XSS];   // staged x tile (16 x 1024 bf16)
    __shared__ __align__(16) __bf16 hs[16 * HSS];   // staged h chunk (16 x 256 bf16)
    __shared__ int   rowtok[16];
    __shared__ float p0s[16];

    int tid = threadIdx.x, lane = tid & 31, w = tid >> 5;
    int m = lane & 15;          // WMMA row (A) or col (B/C) within tile
    int kh = lane >> 4;         // lane half selects K sub-range

    if (tid < 16) {
        int gi = t * 16 + tid;
        bool valid = gi < count;
        int tok = list[e * NTOK + (valid ? gi : 0)];
        rowtok[tid] = tok;
        p0s[tid]    = valid ? p0arr[tok] : 0.0f;   // pad rows contribute 0
    }
    __syncthreads();

    { // stage x rows into LDS: 16 threads/row, 64 bf16 (128B) each
        int r  = tid >> 4;
        int c0 = (tid & 15) * 64;
        const __bf16* g = xbf + (size_t)rowtok[r] * HDIM + c0;
#if HAS_ASYNC_LDS
        // CDNA5 async global->LDS copy (no VGPR bounce), tracked by ASYNCcnt
#pragma unroll
        for (int j = 0; j < 8; ++j)
            __builtin_amdgcn_global_load_async_to_lds_b128(
                (__attribute__((address_space(1))) v4i*)(g + j * 8),
                (__attribute__((address_space(3))) v4i*)&xs[r * XSS + c0 + j * 8],
                0, 0);
        __builtin_amdgcn_s_wait_asynccnt(0);
#else
#pragma unroll
        for (int j = 0; j < 8; ++j)
            *(u32x4*)&xs[r * XSS + c0 + j * 8] = *(const u32x4*)(g + j * 8);
#endif
    }
    __syncthreads();

    v8f acc[8];
    v8f vzero = {0.f, 0.f, 0.f, 0.f, 0.f, 0.f, 0.f, 0.f};
#pragma unroll
    for (int i = 0; i < 8; ++i) acc[i] = vzero;

    const __bf16* xrow = &xs[m * XSS];

    for (int fc = 0; fc < FDIM; fc += 256) {
        // ---- phase 1: h[16, w*32..w*32+32) = gelu(x @ W1 + b1) ----
        int fb = fc + w * 32;
        v8f h0 = vzero, h1 = vzero;
        const __bf16* w1a = W1t + ((size_t)e * FDIM + fb + m) * HDIM;  // col n = fb+m
        const __bf16* w1b = w1a + (size_t)16 * HDIM;                   // col n = fb+m+16
        // prefetch the next chunk's W1 column stream into cache
        if (fc + 256 < FDIM)
            __builtin_prefetch(w1a + (size_t)256 * HDIM, 0, 2);
        for (int k = 0; k < HDIM; k += 32) {
            FragAB a;
            a.u[0] = *(const u32x4*)&xrow[k + kh * 16];
            a.u[1] = *(const u32x4*)&xrow[k + kh * 16 + 8];
            int kb = k + kh * 16;
            FragAB bf0, bf1;
            bf0.u[0] = *(const u32x4*)(w1a + kb);
            bf0.u[1] = *(const u32x4*)(w1a + kb + 8);
            bf1.u[0] = *(const u32x4*)(w1b + kb);
            bf1.u[1] = *(const u32x4*)(w1b + kb + 8);
            h0 = __builtin_amdgcn_wmma_f32_16x16x32_bf16(false, a.v, false, bf0.v,
                                                         (short)0, h0, false, false);
            h1 = __builtin_amdgcn_wmma_f32_16x16x32_bf16(false, a.v, false, bf1.v,
                                                         (short)0, h1, false, false);
        }
        float bia = b1[e * FDIM + fb + m];
        float bib = b1[e * FDIM + fb + m + 16];
        int rb = kh * 8;
        int cl = (fb - fc) + m;
#pragma unroll
        for (int r = 0; r < 8; ++r) {
            hs[(rb + r) * HSS + cl]      = (__bf16)gelu_tanh(h0[r] + bia);
            hs[(rb + r) * HSS + cl + 16] = (__bf16)gelu_tanh(h1[r] + bib);
        }
        __syncthreads();

        // ---- phase 2: acc += h[16,256] @ W2[fc..fc+256, wave cols] ----
        for (int kk = 0; kk < 256; kk += 32) {
            FragAB a;
            a.u[0] = *(const u32x4*)&hs[m * HSS + kk + kh * 16];
            a.u[1] = *(const u32x4*)&hs[m * HSS + kk + kh * 16 + 8];
            int kb2 = fc + kk + kh * 16;
#pragma unroll
            for (int nt = 0; nt < 8; ++nt) {
                const __bf16* wp = W2t + ((size_t)e * HDIM + w * 128 + nt * 16 + m) * FDIM + kb2;
                FragAB bb;
                bb.u[0] = *(const u32x4*)wp;
                bb.u[1] = *(const u32x4*)(wp + 8);
                acc[nt] = __builtin_amdgcn_wmma_f32_16x16x32_bf16(false, a.v, false, bb.v,
                                                                  (short)0, acc[nt], false, false);
            }
        }
        __syncthreads();   // hs reused next iteration
    }

    // ---- epilogue: y = (acc + b2) * p0, scatter-add to output rows ----
#pragma unroll
    for (int nt = 0; nt < 8; ++nt) {
        int col = w * 128 + nt * 16 + m;
        float bias = b2[e * HDIM + col];
#pragma unroll
        for (int r = 0; r < 8; ++r) {
            int row = kh * 8 + r;
            float val = (acc[nt][r] + bias) * p0s[row];
            atomicAdd(&out[(size_t)rowtok[row] * HDIM + col], val);
        }
    }
}

// ---------------- launch ----------------
extern "C" void kernel_launch(void* const* d_in, const int* in_sizes, int n_in,
                              void* d_out, int out_size, void* d_ws, size_t ws_size,
                              hipStream_t stream) {
    const float* x  = (const float*)d_in[0];
    const float* Wr = (const float*)d_in[1];
    const float* br = (const float*)d_in[2];
    const float* W1 = (const float*)d_in[3];
    const float* b1 = (const float*)d_in[4];
    const float* W2 = (const float*)d_in[5];
    const float* b2 = (const float*)d_in[6];
    float* out = (float*)d_out;

    char* ws = (char*)d_ws;
    __bf16* xbf  = (__bf16*)(ws + OFF_XBF);
    __bf16* W1t  = (__bf16*)(ws + OFF_W1T);
    __bf16* W2t  = (__bf16*)(ws + OFF_W2T);
    float*  p0a  = (float*)(ws + OFF_P0);
    int*    cnt  = (int*)(ws + OFF_CNT);
    int*    list = (int*)(ws + OFF_LIST);

    int n4 = NTOK * HDIM / 4;   // 2,097,152
    moe_zero_k<<<(n4 + 255) / 256, 256, 0, stream>>>(out, n4, cnt);
    moe_cvtx_k<<<(n4 + 255) / 256, 256, 0, stream>>>(x, xbf, n4);
    // W1 [E][H][F] -> W1t [E][F][H] (bf16, K=H contiguous per f-column)
    moe_transpose_cvt_k<<<EXP * (HDIM / 32) * (FDIM / 32), 256, 0, stream>>>(W1, W1t, HDIM, FDIM);
    // W2 [E][F][H] -> W2t [E][H][F] (bf16, K=F contiguous per h-column)
    moe_transpose_cvt_k<<<EXP * (FDIM / 32) * (HDIM / 32), 256, 0, stream>>>(W2, W2t, FDIM, HDIM);
    moe_router_k<<<NTOK / 8, 256, 0, stream>>>(x, Wr, br, p0a, cnt, list);
    moe_ffn_k<<<EXP * 512, 256, 0, stream>>>(xbf, W1t, W2t, b1, b2, p0a, cnt, list, out);
}
